// Attention_51290499449352
// MI455X (gfx1250) — matrix operand order repaired and checked by reading, not compiled, not run on hardware
//
#include <hip/hip_runtime.h>
#include <hip/hip_bf16.h>

// ---------------------------------------------------------------------------
// Attention block for MI455X (gfx1250): bf16 WMMA everywhere, fp32 accumulate.
// Weights pre-transposed (n-major) so B-tiles stage with b128 ops only.
// V written head-transposed [B,H,D,N] so PV B-fragments load straight from
// global memory. GEMM staging uses GLOBAL_LOAD_ASYNC_TO_LDS (ASYNCcnt DMA).
// ---------------------------------------------------------------------------

typedef __bf16 bf16;
typedef __attribute__((ext_vector_type(8)))  bf16  v8bf;
typedef __attribute__((ext_vector_type(16))) bf16  v16bf;
typedef __attribute__((ext_vector_type(8)))  float v8f;

#define DIM    768
#define NHEAD  12
#define HDIM   64
#define SEQ    2048
#define BATCH  2
#define MROWS  (BATCH * SEQ)   // 4096
#define NQKV   (3 * DIM)       // 2304
#define ATTN_SCALE 0.125f      // 1/sqrt(64)

#if defined(__AMDGCN__) && __has_builtin(__builtin_amdgcn_global_load_async_to_lds_b128)
#define USE_ASYNC_LDS 1
#else
#define USE_ASYNC_LDS 0
#endif

#if USE_ASYNC_LDS
// Exact parameter types per hipcc diagnostic:
//   arg0: int __vector(4) __device__(AS1) *   (global source)
//   arg1: LDS destination (AS3)
typedef int v4i __attribute__((vector_size(16)));
typedef __attribute__((address_space(1))) v4i* gv4i_p;
typedef __attribute__((address_space(3))) v4i* lv4i_p;
// 16B global -> LDS async DMA (tracked by ASYNCcnt, no VGPR round-trip).
static __device__ __forceinline__ void async_copy16(const bf16* g, bf16* l) {
    __builtin_amdgcn_global_load_async_to_lds_b128(
        (gv4i_p)(v4i*)(void*)g, (lv4i_p)(v4i*)(void*)l, /*offset=*/0, /*cpol=*/0);
}
static __device__ __forceinline__ void wait_async0() {
#if __has_builtin(__builtin_amdgcn_s_wait_asynccnt)
    __builtin_amdgcn_s_wait_asynccnt(0);
#else
    asm volatile("s_wait_asynccnt 0" ::: "memory");
#endif
}
#endif

// Build a v16bf from two contiguous 8-element (16B) chunks.
static __device__ __forceinline__ v16bf frag_2x8(const bf16* p0, const bf16* p1) {
    v8bf a = *(const v8bf*)p0;
    v8bf b = *(const v8bf*)p1;
    v16bf r;
#pragma unroll
    for (int i = 0; i < 8; ++i) { r[i] = a[i]; r[i + 8] = b[i]; }
    return r;
}

// A-fragment (16x32 bf16) from a row-major tile with leading dim `ld`.
// Lanes 0-15 hold row m=lane, K 0..15 (contiguous, matching B);
//          lanes 16-31 hold row m=lane-16, K 16..31.
static __device__ __forceinline__ v16bf load_a_frag(const bf16* base, int ld, int lane) {
    int m  = lane & 15;
    int k0 = (lane >> 4) * 16;
    const bf16* p = base + m * ld + k0;
    return frag_2x8(p, p + 8);
}

// B-fragment (32x16 bf16) from an n-major (transposed) tile with leading dim
// `ld`. lane = column n (mod 16); this code's k order: half-wave selects K block 0..15 /
// 16..31, contiguous in K -> single 32B read.
static __device__ __forceinline__ v16bf load_b_frag(const bf16* bt, int ld, int lane) {
    int n  = lane & 15;
    int ks = (lane >> 4) * 16;
    return *(const v16bf*)(bt + n * ld + ks);
}

static __device__ __forceinline__ v8f wmma_bf16(v16bf a, v16bf b, v8f c) {
    return __builtin_amdgcn_wmma_f32_16x16x32_bf16(
        /*neg_a=*/false, a, /*neg_b=*/false, b,
        /*c_mod=*/(short)0, c, /*reuse_a=*/false, /*reuse_b=*/false);
}

// ---------------------------------------------------------------------------
// fp32 -> bf16 conversion (elementwise)
// ---------------------------------------------------------------------------
__global__ void cvt_f32_to_bf16(const float* __restrict__ in,
                                bf16* __restrict__ out, int n) {
    int i = blockIdx.x * blockDim.x + threadIdx.x;
    int stride = gridDim.x * blockDim.x;
    for (; i < n; i += stride) out[i] = (bf16)in[i];
}

// fp32 [K][N] -> bf16 [N][K] transpose-convert (one-time weight staging)
__global__ void cvt_f32_to_bf16_t(const float* __restrict__ in,
                                  bf16* __restrict__ out, int K, int N) {
    int i = blockIdx.x * blockDim.x + threadIdx.x;
    int stride = gridDim.x * blockDim.x;
    int total = K * N;
    for (; i < total; i += stride) {
        int k = i / N, n = i - k * N;
        out[(size_t)n * K + k] = (bf16)in[i];
    }
}

// ---------------------------------------------------------------------------
// Shared GEMM mainloop: block tile 128x64, K-step 64, 256 threads = 8 waves
// in a 4x2 grid; each wave owns a 32x32 output tile (2x2 WMMA accumulators,
// 8 WMMAs per wave per barrier pair). B is n-major in global memory (pre-transposed).
// ---------------------------------------------------------------------------
template<int LDA, int KDIM>
static __device__ __forceinline__ void gemm_mainloop(
    const bf16* __restrict__ A, const bf16* __restrict__ Bt,  // Bt: [N][KDIM]
    int rowBase, int colBase, int tid,
    v8f c[2][2], bf16 (*As)[64], bf16 (*Bts)[64]) {
    int lane = tid & 31;
    int w    = tid >> 5;
    int wm   = w >> 1, wn = w & 1;          // 4 x 2 wave grid
    for (int k0 = 0; k0 < KDIM; k0 += 64) {
        int ra = tid >> 1, h0 = (tid & 1) * 32;   // A: row, 32-elem half
        int rb = tid >> 2, s0 = (tid & 3) * 16;   // B: row, 16-elem quarter
        const bf16* srcA = &A[(size_t)(rowBase + ra) * LDA + k0 + h0];
        const bf16* srcB = &Bt[(size_t)(colBase + rb) * KDIM + k0 + s0];
#if USE_ASYNC_LDS
        // DMA global -> LDS, no VGPR round-trip (ASYNCcnt tracked).
        async_copy16(srcA,      &As[ra][h0]);
        async_copy16(srcA + 8,  &As[ra][h0 + 8]);
        async_copy16(srcA + 16, &As[ra][h0 + 16]);
        async_copy16(srcA + 24, &As[ra][h0 + 24]);
        async_copy16(srcB,      &Bts[rb][s0]);
        async_copy16(srcB + 8,  &Bts[rb][s0 + 8]);
#else
        {
            v8bf t0 = *(const v8bf*)(srcA);
            v8bf t1 = *(const v8bf*)(srcA + 8);
            v8bf t2 = *(const v8bf*)(srcA + 16);
            v8bf t3 = *(const v8bf*)(srcA + 24);
            *(v8bf*)&As[ra][h0]      = t0;
            *(v8bf*)&As[ra][h0 + 8]  = t1;
            *(v8bf*)&As[ra][h0 + 16] = t2;
            *(v8bf*)&As[ra][h0 + 24] = t3;
            v8bf u0 = *(const v8bf*)(srcB);
            v8bf u1 = *(const v8bf*)(srcB + 8);
            *(v8bf*)&Bts[rb][s0]     = u0;
            *(v8bf*)&Bts[rb][s0 + 8] = u1;
        }
#endif
        if (k0 + 64 < KDIM) {
            __builtin_prefetch(srcA + 64, 0, 3);   // near-scope prefetch
            __builtin_prefetch(srcB + 64, 0, 3);
        }
#if USE_ASYNC_LDS
        wait_async0();     // own transfers landed; barrier covers the rest
#endif
        __syncthreads();
#pragma unroll
        for (int ks = 0; ks < 2; ++ks) {
#pragma unroll
            for (int mi = 0; mi < 2; ++mi) {
                v16bf a = load_a_frag(&As[wm * 32 + mi * 16][ks * 32], 64, lane);
#pragma unroll
                for (int ni = 0; ni < 2; ++ni) {
                    v16bf b = load_b_frag(&Bts[wn * 32 + ni * 16][ks * 32], 64, lane);
                    c[mi][ni] = wmma_bf16(a, b, c[mi][ni]);
                }
            }
        }
        __syncthreads();
    }
}

// ---------------------------------------------------------------------------
// Kernel: QKV = Xb @ Wqkv^T' + b, scattered into Q/K [B,H,N,D] and
// V transposed [B,H,D,N] (bf16). grid (NQKV/64, MROWS/128), block 256.
// ---------------------------------------------------------------------------
__global__ __launch_bounds__(256)
void qkv_gemm_kernel(const bf16* __restrict__ X, const bf16* __restrict__ Wt,
                     const float* __restrict__ bias,
                     bf16* __restrict__ Q, bf16* __restrict__ K,
                     bf16* __restrict__ Vt) {
    __shared__ bf16 As[128][64];
    __shared__ bf16 Bts[64][64];
    int tid = threadIdx.x, lane = tid & 31, w = tid >> 5;
    int wm = w >> 1, wn = w & 1;
    int rowBase = blockIdx.y * 128, colBase = blockIdx.x * 64;

    v8f c[2][2] = {};
    gemm_mainloop<DIM, DIM>(X, Wt, rowBase, colBase, tid, c, As, Bts);

    // Block-uniform Q/K/V segment: 64-wide tiles never straddle (64 | 768).
    int which   = colBase / DIM;
    int remBase = colBase - which * DIM;
    int sel = lane >> 4, nn = lane & 15;
#pragma unroll
    for (int mi = 0; mi < 2; ++mi) {
#pragma unroll
        for (int ni = 0; ni < 2; ++ni) {
            int rem = remBase + wn * 32 + ni * 16 + nn;
            int h = rem >> 6, d = rem & 63;
            float bv = bias[which * DIM + rem];
            if (which == 2) {
#pragma unroll
                for (int r = 0; r < 8; ++r) {
                    int grow = rowBase + wm * 32 + mi * 16 + r + 8 * sel;
                    int b = grow >> 11, n = grow & (SEQ - 1);
                    Vt[(size_t)((b * NHEAD + h) * HDIM + d) * SEQ + n] =
                        (bf16)(c[mi][ni][r] + bv);
                }
            } else {
                bf16* dst = (which == 0) ? Q : K;
#pragma unroll
                for (int r = 0; r < 8; ++r) {
                    int grow = rowBase + wm * 32 + mi * 16 + r + 8 * sel;
                    int b = grow >> 11, n = grow & (SEQ - 1);
                    dst[(size_t)((b * NHEAD + h) * SEQ + n) * HDIM + d] =
                        (bf16)(c[mi][ni][r] + bv);
                }
            }
        }
    }
}

// ---------------------------------------------------------------------------
// Kernel: flash attention. Block = 64 queries of one (b,h); 4 waves, each
// owns 16 query rows. Streams keys 64 at a time with online softmax
// (16 WMMAs per iteration). No block barriers: K and V^T B-fragments load
// directly from global; only the wave-private prob transpose touches LDS.
// grid (SEQ/64, NHEAD, BATCH), block 128.
// ---------------------------------------------------------------------------
__global__ __launch_bounds__(128)
void attn_kernel(const bf16* __restrict__ Q, const bf16* __restrict__ K,
                 const bf16* __restrict__ Vt, bf16* __restrict__ O) {
    __shared__ bf16 Pt[4][16][64];   // per-wave prob tile (C-layout -> A-layout)
    int tid = threadIdx.x, lane = tid & 31, w = tid >> 5;
    int sel = lane >> 4, nn = lane & 15;
    int h = blockIdx.y, b = blockIdx.z;
    int bh = b * NHEAD + h;
    const bf16* Qh  = Q  + (size_t)bh * SEQ * HDIM;
    const bf16* Kh  = K  + (size_t)bh * SEQ * HDIM;
    const bf16* Vth = Vt + (size_t)bh * HDIM * SEQ;   // [d][key]
    int qbase = blockIdx.x * 64 + w * 16;

    // Q A-fragments (16 x 64 head dim = two K=32 fragments), VGPR-resident.
    v16bf aq0 = load_a_frag(Qh + (size_t)qbase * HDIM, HDIM, lane);
    v16bf aq1 = load_a_frag(Qh + (size_t)qbase * HDIM + 32, HDIM, lane);

    v8f o[4] = {};                   // 16 x 64 fp32 output accumulator
    float mstat[8], lstat[8];
#pragma unroll
    for (int r = 0; r < 8; ++r) { mstat[r] = -1e30f; lstat[r] = 0.0f; }

    for (int j = 0; j < SEQ; j += 64) {
        // Scores: S = Q @ K^T for four 16-key tiles; K is [key][d] row-major,
        // which is exactly n-major for the B operand -> 32B contiguous reads.
        v8f s[4];
#pragma unroll
        for (int t = 0; t < 4; ++t) {
            int key = j + t * 16 + nn;
            const bf16* kp = &Kh[(size_t)key * HDIM + sel * 16];
            v16bf bk0 = *(const v16bf*)(kp);
            v16bf bk1 = *(const v16bf*)(kp + 32);
            v8f z = {};
            z = wmma_bf16(aq0, bk0, z);
            z = wmma_bf16(aq1, bk1, z);
            s[t] = z;
        }
        if (j + 64 < SEQ) {
            __builtin_prefetch(&Kh[(size_t)(j + 64 + nn) * HDIM], 0, 3);
            __builtin_prefetch(&Vth[(size_t)nn * SEQ + j + 64], 0, 3);
        }
        // Online softmax. C-layout: VGPR r holds row (r + 8*sel); the 16 lanes
        // of each half hold the key columns -> xor-reduce within halves.
        float alpha[8];
#pragma unroll
        for (int r = 0; r < 8; ++r) {
#pragma unroll
            for (int t = 0; t < 4; ++t) s[t][r] *= ATTN_SCALE;
            float v = fmaxf(fmaxf(s[0][r], s[1][r]), fmaxf(s[2][r], s[3][r]));
#pragma unroll
            for (int mk = 1; mk < 16; mk <<= 1)
                v = fmaxf(v, __shfl_xor(v, mk, 32));
            float mc = fmaxf(mstat[r], v);
            alpha[r] = __expf(mstat[r] - mc);
            mstat[r] = mc;
            float rs = 0.0f;
#pragma unroll
            for (int t = 0; t < 4; ++t) {
                float p = __expf(s[t][r] - mc);
                s[t][r] = p;
                rs += p;
            }
#pragma unroll
            for (int mk = 1; mk < 16; mk <<= 1)
                rs += __shfl_xor(rs, mk, 32);
            lstat[r] = lstat[r] * alpha[r] + rs;
        }
        // Rescale O accumulators, stash P tile into wave-private LDS region.
        // (LDS ops are in-order within a wave; compiler inserts dscnt waits.)
#pragma unroll
        for (int r = 0; r < 8; ++r) {
            int m = r + 8 * sel;
#pragma unroll
            for (int t = 0; t < 4; ++t) Pt[w][m][t * 16 + nn] = (bf16)s[t][r];
#pragma unroll
            for (int t = 0; t < 4; ++t) o[t][r] *= alpha[r];
        }
        // O += P @ V : A = probs 16x64 (two K=32 fragments via LDS reshape),
        // B = V^T tiles straight from global.
        v16bf ap0 = load_a_frag(&Pt[w][0][0], 64, lane);
        v16bf ap1 = load_a_frag(&Pt[w][0][32], 64, lane);
#pragma unroll
        for (int t = 0; t < 4; ++t) {
            const bf16* vp = &Vth[(size_t)(t * 16 + nn) * SEQ + j + sel * 16];
            v16bf bv0 = *(const v16bf*)(vp);
            v16bf bv1 = *(const v16bf*)(vp + 32);
            o[t] = wmma_bf16(ap0, bv0, o[t]);
            o[t] = wmma_bf16(ap1, bv1, o[t]);
        }
    }
    // Normalize and store bf16 into O [B, N, C] (= [B, N, H*D]).
#pragma unroll
    for (int t = 0; t < 4; ++t) {
        int col = h * HDIM + t * 16 + nn;
#pragma unroll
        for (int r = 0; r < 8; ++r) {
            int grow = qbase + r + 8 * sel;
            O[(size_t)(b * SEQ + grow) * DIM + col] = (bf16)(o[t][r] / lstat[r]);
        }
    }
}

// ---------------------------------------------------------------------------
// Kernel: Y = Ob @ Wproj^T' + b_proj, fp32 output.
// grid (DIM/64, MROWS/128), block 256.
// ---------------------------------------------------------------------------
__global__ __launch_bounds__(256)
void proj_gemm_kernel(const bf16* __restrict__ A, const bf16* __restrict__ Wt,
                      const float* __restrict__ bias, float* __restrict__ out) {
    __shared__ bf16 As[128][64];
    __shared__ bf16 Bts[64][64];
    int tid = threadIdx.x, lane = tid & 31, w = tid >> 5;
    int wm = w >> 1, wn = w & 1;
    int rowBase = blockIdx.y * 128, colBase = blockIdx.x * 64;

    v8f c[2][2] = {};
    gemm_mainloop<DIM, DIM>(A, Wt, rowBase, colBase, tid, c, As, Bts);

    int sel = lane >> 4, nn = lane & 15;
#pragma unroll
    for (int mi = 0; mi < 2; ++mi) {
#pragma unroll
        for (int ni = 0; ni < 2; ++ni) {
            int gc = colBase + wn * 32 + ni * 16 + nn;
            float bv = bias[gc];
#pragma unroll
            for (int r = 0; r < 8; ++r) {
                int grow = rowBase + wm * 32 + mi * 16 + r + 8 * sel;
                out[(size_t)grow * DIM + gc] = c[mi][ni][r] + bv;
            }
        }
    }
}

// ---------------------------------------------------------------------------
// Host launcher
// ---------------------------------------------------------------------------
extern "C" void kernel_launch(void* const* d_in, const int* in_sizes, int n_in,
                              void* d_out, int out_size, void* d_ws, size_t ws_size,
                              hipStream_t stream) {
    const float* x      = (const float*)d_in[0];   // [2,2048,768]
    const float* w_qkv  = (const float*)d_in[1];   // [768,2304]
    const float* b_qkv  = (const float*)d_in[2];   // [2304]
    const float* w_proj = (const float*)d_in[3];   // [768,768]
    const float* b_proj = (const float*)d_in[4];   // [768]
    float* out = (float*)d_out;

    // Workspace carve-up (bf16 staging, 256B-aligned).
    size_t off = 0;
    auto carve = [&](size_t bytes) {
        void* p = (char*)d_ws + off;
        off += (bytes + 255) & ~(size_t)255;
        return p;
    };
    bf16* Xb   = (bf16*)carve((size_t)MROWS * DIM * sizeof(bf16));
    bf16* Wqt  = (bf16*)carve((size_t)NQKV * DIM * sizeof(bf16));   // [2304][768]
    bf16* Wpt  = (bf16*)carve((size_t)DIM * DIM * sizeof(bf16));    // [768][768]
    bf16* Qb   = (bf16*)carve((size_t)BATCH * NHEAD * SEQ * HDIM * sizeof(bf16));
    bf16* Kb   = (bf16*)carve((size_t)BATCH * NHEAD * SEQ * HDIM * sizeof(bf16));
    bf16* Vtb  = (bf16*)carve((size_t)BATCH * NHEAD * HDIM * SEQ * sizeof(bf16));
    bf16* Ob   = (bf16*)carve((size_t)MROWS * DIM * sizeof(bf16));
    (void)ws_size; (void)in_sizes; (void)n_in; (void)out_size;

    // 1) Staging: x elementwise; weights transpose-converted to n-major.
    cvt_f32_to_bf16<<<2048, 256, 0, stream>>>(x, Xb, MROWS * DIM);
    cvt_f32_to_bf16_t<<<2048, 256, 0, stream>>>(w_qkv,  Wqt, DIM, NQKV);
    cvt_f32_to_bf16_t<<<1024, 256, 0, stream>>>(w_proj, Wpt, DIM, DIM);

    // 2) QKV projection (fused bias; Q,K head-major, V head-transposed)
    qkv_gemm_kernel<<<dim3(NQKV / 64, MROWS / 128), 256, 0, stream>>>(
        Xb, Wqt, b_qkv, Qb, Kb, Vtb);

    // 3) Flash attention (barrier-free inner loop, 64-key blocks)
    attn_kernel<<<dim3(SEQ / 64, NHEAD, BATCH), 128, 0, stream>>>(Qb, Kb, Vtb, Ob);

    // 4) Output projection (fp32 result)
    proj_gemm_kernel<<<dim3(DIM / 64, MROWS / 128), 256, 0, stream>>>(
        Ob, Wpt, b_proj, out);
}
